// DKVMN_62302795596377
// MI455X (gfx1250) — compile-verified, matrix operand order checked by reading, not run on hardware
//
#include <hip/hip_runtime.h>

typedef __attribute__((ext_vector_type(16))) __bf16 v16bf;
typedef __attribute__((ext_vector_type(8)))  float  v8f;

#define B_  2048
#define M_  512
#define KD_ 128
#define VD_ 256

// ---------- helpers -------------------------------------------------------

__device__ inline unsigned short f2bf(float f) {
  union { float f; unsigned u; } x; x.f = f;
  unsigned r = x.u + 0x7FFFu + ((x.u >> 16) & 1u);   // round-to-nearest-even
  return (unsigned short)(r >> 16);
}

// Build a 16x32 bf16 WMMA fragment (A layout, ISA 7.12.2) from a row-major
// f32 matrix. Same loader serves B fragments because B[k][n] = W[n][k] when
// W is row-major: lane's "row" is then the output column n.
// Per-lane element e holds K = kb + e + (e>=8?8:0) + (lane>=16?8:0).
__device__ inline v16bf load_frag_f32(const float* __restrict__ p, int ld,
                                      int row0, int kb) {
  int lane = threadIdx.x & 31;
  int r    = row0 + (lane & 15);
  int koff = kb + ((lane >> 4) << 3);
  const float4* s = (const float4*)(p + (size_t)r * ld + koff);
  float4 f0 = s[0], f1 = s[1], f2 = s[4], f3 = s[5];
  union { v16bf v; unsigned short h[16]; } u;
  u.h[0]=f2bf(f0.x);  u.h[1]=f2bf(f0.y);  u.h[2]=f2bf(f0.z);  u.h[3]=f2bf(f0.w);
  u.h[4]=f2bf(f1.x);  u.h[5]=f2bf(f1.y);  u.h[6]=f2bf(f1.z);  u.h[7]=f2bf(f1.w);
  u.h[8]=f2bf(f2.x);  u.h[9]=f2bf(f2.y);  u.h[10]=f2bf(f2.z); u.h[11]=f2bf(f2.w);
  u.h[12]=f2bf(f3.x); u.h[13]=f2bf(f3.y); u.h[14]=f2bf(f3.z); u.h[15]=f2bf(f3.w);
  return u.v;
}

// Same fragment from an already-bf16 row-major matrix (two b128 loads).
__device__ inline v16bf load_frag_bf16(const unsigned short* __restrict__ p,
                                       int ld, int row0, int kb) {
  int lane = threadIdx.x & 31;
  int r    = row0 + (lane & 15);
  int koff = kb + ((lane >> 4) << 3);
  const unsigned short* s = p + (size_t)r * ld + koff;
  union { v16bf v; uint4 q[2]; } u;
  u.q[0] = *(const uint4*)(s);
  u.q[1] = *(const uint4*)(s + 16);
  return u.v;
}

__device__ inline v8f wmma_bf16(v16bf a, v16bf b, v8f c) {
  return __builtin_amdgcn_wmma_f32_16x16x32_bf16(
      /*neg_a=*/false, a, /*neg_b=*/false, b,
      /*c_mod=*/(short)0, c, /*reuse_a=*/false, /*reuse_b=*/false);
}

// ---------- K1: qk = tanh(q @ Wq^T + bq), stored bf16 --------------------

__global__ __launch_bounds__(32)
void k_query(const float* __restrict__ q, const float* __restrict__ Wq,
             const float* __restrict__ bq, unsigned short* __restrict__ qk_bf) {
  int b0 = blockIdx.x * 16;
  int lane = threadIdx.x;
  v16bf a[4];
#pragma unroll
  for (int ks = 0; ks < 4; ++ks) a[ks] = load_frag_f32(q, KD_, b0, ks * 32);
#pragma unroll
  for (int nt = 0; nt < 8; ++nt) {
    int n0 = nt * 16;
    v8f acc = {};
#pragma unroll
    for (int ks = 0; ks < 4; ++ks)
      acc = wmma_bf16(a[ks], load_frag_f32(Wq, KD_, n0, ks * 32), acc);
    int n = n0 + (lane & 15);
    float bias = bq[n];
#pragma unroll
    for (int i = 0; i < 8; ++i) {
      int m = i + ((lane >> 4) << 3);        // C/D layout: VGPR i -> row
      qk_bf[(size_t)(b0 + m) * KD_ + n] = f2bf(tanhf(acc[i] + bias));
    }
  }
}

// ---------- K2: logits = qk @ Km^T ; softmax over M -> corr --------------

__global__ __launch_bounds__(32)
void k_attn(const unsigned short* __restrict__ qk_bf,
            const float* __restrict__ Km, float* __restrict__ corr) {
  __shared__ float lg[16 * M_];              // 32 KB logits tile
  int b0 = blockIdx.x * 16;
  int lane = threadIdx.x;
  v16bf a[4];
#pragma unroll
  for (int ks = 0; ks < 4; ++ks) a[ks] = load_frag_bf16(qk_bf, KD_, b0, ks * 32);
  for (int nt = 0; nt < 32; ++nt) {
    int n0 = nt * 16;
    v8f acc = {};
#pragma unroll
    for (int ks = 0; ks < 4; ++ks)
      acc = wmma_bf16(a[ks], load_frag_f32(Km, KD_, n0, ks * 32), acc);
    int n = n0 + (lane & 15);
#pragma unroll
    for (int i = 0; i < 8; ++i) {
      int m = i + ((lane >> 4) << 3);
      lg[m * M_ + n] = acc[i];
    }
  }
  __syncthreads();
  // per-row softmax, wave32 cooperative
  for (int r = 0; r < 16; ++r) {
    float mx = -1e30f;
    for (int j = lane; j < M_; j += 32) mx = fmaxf(mx, lg[r * M_ + j]);
#pragma unroll
    for (int off = 16; off; off >>= 1) mx = fmaxf(mx, __shfl_xor(mx, off));
    float sm = 0.f;
    for (int j = lane; j < M_; j += 32) {
      float e = __expf(lg[r * M_ + j] - mx);
      lg[r * M_ + j] = e;
      sm += e;
    }
#pragma unroll
    for (int off = 16; off; off >>= 1) sm += __shfl_xor(sm, off);
    float inv = 1.f / sm;
    for (int j = lane; j < M_; j += 32)
      corr[(size_t)(b0 + r) * M_ + j] = lg[r * M_ + j] * inv;
  }
}

// ---------- K3: erase = sigmoid(v@We^T+be), add = tanh(v@Wa^T+ba) --------

__global__ __launch_bounds__(32)
void k_erase_add(const float* __restrict__ v,
                 const float* __restrict__ We, const float* __restrict__ be,
                 const float* __restrict__ Wa, const float* __restrict__ ba,
                 float* __restrict__ erase, float* __restrict__ addv) {
  int b0 = blockIdx.x * 16;
  int lane = threadIdx.x;
  v16bf a[8];
#pragma unroll
  for (int ks = 0; ks < 8; ++ks) a[ks] = load_frag_f32(v, VD_, b0, ks * 32);
  for (int nt = 0; nt < 16; ++nt) {
    int n0 = nt * 16;
    v8f ae = {}, aa = {};
#pragma unroll
    for (int ks = 0; ks < 8; ++ks)
      ae = wmma_bf16(a[ks], load_frag_f32(We, VD_, n0, ks * 32), ae);
#pragma unroll
    for (int ks = 0; ks < 8; ++ks)
      aa = wmma_bf16(a[ks], load_frag_f32(Wa, VD_, n0, ks * 32), aa);
    int n = n0 + (lane & 15);
    float b1 = be[n], b2 = ba[n];
#pragma unroll
    for (int i = 0; i < 8; ++i) {
      int m = i + ((lane >> 4) << 3);
      size_t idx = (size_t)(b0 + m) * VD_ + n;
      erase[idx] = 1.f / (1.f + __expf(-(ae[i] + b1)));
      addv[idx]  = tanhf(aa[i] + b2);
    }
  }
}

// ---------- K4: fused streaming pass (bandwidth-critical) ----------------
// One block per batch row: reads Vm[b] once (512 KB), writes updated[b]
// once, and accumulates read_content[b] on the fly. All b128 traffic.

__global__ __launch_bounds__(256)
void k_update(const float* __restrict__ corr, const float* __restrict__ erase,
              const float* __restrict__ addv, const float* __restrict__ Vm,
              float* __restrict__ rc, float* __restrict__ upd) {
  int b = blockIdx.x;
  int tid = threadIdx.x;
  int g = tid >> 6;          // m-phase 0..3
  int c = tid & 63;          // float4 column group (64*4 = 256 = VD)
  __shared__ float  scorr[M_];
  __shared__ float4 red[256];
  for (int j = tid; j < M_; j += 256) scorr[j] = corr[(size_t)b * M_ + j];
  float4 er = ((const float4*)(erase + (size_t)b * VD_))[c];
  float4 ad = ((const float4*)(addv  + (size_t)b * VD_))[c];
  __syncthreads();
  const float4* Vp = (const float4*)(Vm + (size_t)b * M_ * VD_);
  float4*       Up = (float4*)(upd + (size_t)b * M_ * VD_);
  float4 acc = {0.f, 0.f, 0.f, 0.f};
  for (int m = g; m < M_; m += 4) {
    __builtin_prefetch(&Vp[(m + 16) * (VD_ / 4) + c], 0, 0);
    float cm = scorr[m];
    float4 val = Vp[m * (VD_ / 4) + c];
    acc.x = fmaf(cm, val.x, acc.x);
    acc.y = fmaf(cm, val.y, acc.y);
    acc.z = fmaf(cm, val.z, acc.z);
    acc.w = fmaf(cm, val.w, acc.w);
    float4 o;
    o.x = fmaf(val.x, 1.f - cm * er.x, cm * ad.x);
    o.y = fmaf(val.y, 1.f - cm * er.y, cm * ad.y);
    o.z = fmaf(val.z, 1.f - cm * er.z, cm * ad.z);
    o.w = fmaf(val.w, 1.f - cm * er.w, cm * ad.w);
    Up[m * (VD_ / 4) + c] = o;
  }
  red[tid] = acc;
  __syncthreads();
  if (g == 0) {
    float4 r1 = red[tid + 64], r2 = red[tid + 128], r3 = red[tid + 192];
    acc.x += r1.x + r2.x + r3.x;
    acc.y += r1.y + r2.y + r3.y;
    acc.z += r1.z + r2.z + r3.z;
    acc.w += r1.w + r2.w + r3.w;
    ((float4*)(rc + (size_t)b * VD_))[c] = acc;
  }
}

// ---------- launch -------------------------------------------------------

extern "C" void kernel_launch(void* const* d_in, const int* in_sizes, int n_in,
                              void* d_out, int out_size, void* d_ws, size_t ws_size,
                              hipStream_t stream) {
  const float* q   = (const float*)d_in[0];   // query_embedded  (B,KD)
  const float* ve  = (const float*)d_in[1];   // value_embedded  (B,VD)
  const float* Vm  = (const float*)d_in[2];   // value_memory    (B,M,VD)
  const float* Km  = (const float*)d_in[3];   // key_memory      (M,KD)
  const float* Wq  = (const float*)d_in[4];
  const float* bq  = (const float*)d_in[5];
  const float* We  = (const float*)d_in[6];
  const float* be  = (const float*)d_in[7];
  const float* Wa  = (const float*)d_in[8];
  const float* ba  = (const float*)d_in[9];

  float* rc  = (float*)d_out;                     // (B,VD)
  float* upd = rc + (size_t)B_ * VD_;             // (B,M,VD)

  char* w = (char*)d_ws;
  unsigned short* qk_bf = (unsigned short*)w;                     // 512 KB
  float* corr  = (float*)(w + (size_t)B_ * KD_ * sizeof(unsigned short));
  float* erase = corr + (size_t)B_ * M_;                          // 4 MB after corr
  float* addv  = erase + (size_t)B_ * VD_;                        // +2 MB

  k_query   <<<B_ / 16, 32, 0, stream>>>(q, Wq, bq, qk_bf);
  k_attn    <<<B_ / 16, 32, 0, stream>>>(qk_bf, Km, corr);
  k_erase_add<<<B_ / 16, 32, 0, stream>>>(ve, We, be, Wa, ba, erase, addv);
  k_update  <<<B_, 256, 0, stream>>>(corr, erase, addv, Vm, rc, upd);
}
